// GAT_91001767068072
// MI455X (gfx1250) — matrix operand order repaired and checked by reading, not compiled, not run on hardware
//
#include <hip/hip_runtime.h>
#include <hip/hip_bf16.h>
#include <math.h>

// ---------------------------------------------------------------------------
// Types for CDNA5 WMMA (wave32): v_wmma_f32_16x16x32_bf16
// ---------------------------------------------------------------------------
typedef __attribute__((ext_vector_type(16))) __bf16 v16bf;
typedef __attribute__((ext_vector_type(8)))  float  v8f;

// Monotonic float <-> uint mapping for atomic max over floats
__device__ __forceinline__ unsigned floatOrd(float f) {
  unsigned u = __float_as_uint(f);
  return (u & 0x80000000u) ? ~u : (u | 0x80000000u);
}
__device__ __forceinline__ float ordFloat(unsigned o) {
  return (o & 0x80000000u) ? __uint_as_float(o & 0x7fffffffu)
                           : __uint_as_float(~o);
}
#define ORD_NEG_INF 0x007FFFFFu   // floatOrd(-inf)

// ---------------------------------------------------------------------------
// Pack weight matrix W[K,N] (row-major f32) into WMMA B-fragment order, bf16.
// Fragment (kt,nt) = 512 bf16: lane l (0..31), elem e (0..15):
//   K = kt*32 + (l>>4)*8 + (e&7) + (e>>3)*16 ,  N = nt*16 + (l&15)
// (16-bit B 32x16 VGPR layout: elems 0..7 = K 8h..8h+7, 8..15 = 16+8h..+7
//  -> the hot-loop B-load is one contiguous 32B v16bf)
// ---------------------------------------------------------------------------
__global__ void pack_w_bf16(const float* __restrict__ W, int K, int N,
                            int Kp, int Np, __bf16* __restrict__ out) {
  int idx = blockIdx.x * blockDim.x + threadIdx.x;
  int ntiles = Np >> 4;
  int total = (Kp >> 5) * ntiles * 512;
  if (idx >= total) return;
  int frag   = idx >> 9;
  int within = idx & 511;
  int lane = within >> 4;
  int e    = within & 15;
  int nt = frag % ntiles;
  int kt = frag / ntiles;
  int k = (kt << 5) + ((lane >> 4) << 3) + (e & 7) + ((e >> 3) << 4);
  int n = (nt << 4) + (lane & 15);
  float v = (k < K && n < N) ? W[(size_t)k * N + n] : 0.0f;
  out[idx] = (__bf16)v;                 // native v_cvt (RTNE)
}

// ---------------------------------------------------------------------------
// WMMA GEMM: C[M, NTILES*16] = A[M, K(lda)] x Bpacked.
//  - 8 waves / block; packed B (KTILES*NTILES*512 bf16) staged once per block
//    in LDS, inner-loop B reads are ds_load_b128 pairs.
//  - One wave per 16-row strip computes ALL column tiles -> A streamed once.
//  - Branch-free inner loop over the full k-tiles; partial tail hoisted out.
// A-fragment per the 16-bit A 16x32 ISA layout:
//   row = lane&15, khalf = lane>>4; elems 0..7 = K kb..kb+7, 8..15 = kb+16..+23
// ---------------------------------------------------------------------------
template <int NTILES, int KTILES>
__global__ void __launch_bounds__(256)
wmma_gemm_bf16(const float* __restrict__ A, const __bf16* __restrict__ Bpk,
               float* __restrict__ C, int M, int K, int lda) {
  constexpr int BELEMS = KTILES * NTILES * 512;     // bf16 elements
  __shared__ __bf16 Bs[BELEMS];

  // cooperative stage of packed B into LDS (16B per thread per step)
  {
    const int4* g = (const int4*)Bpk;
    int4* l = (int4*)Bs;
    for (int i = threadIdx.x; i < BELEMS / 8; i += blockDim.x) l[i] = g[i];
  }
  __syncthreads();                       // before any wave-level exit

  const int wave = blockIdx.x * 8 + (int)(threadIdx.x >> 5);
  const int lane = threadIdx.x & 31;
  const int m0 = wave << 4;
  if (m0 >= M) return;                   // whole-wave exit: EXEC uniform
  const int r  = lane & 15;
  const int kh = lane >> 4;
  const float* row = A + (size_t)(m0 + r) * lda;

  const v8f vzero = {0.f, 0.f, 0.f, 0.f, 0.f, 0.f, 0.f, 0.f};
  v8f acc[NTILES];
#pragma unroll
  for (int t = 0; t < NTILES; ++t) acc[t] = vzero;

  const int kfull = K >> 5;              // fully in-range k-tiles
  for (int kt = 0; kt < kfull; ++kt) {
    const int kb = (kt << 5) + (kh << 3);
    float4 q0 = *(const float4*)(row + kb);
    float4 q1 = *(const float4*)(row + kb + 4);
    float4 q2 = *(const float4*)(row + kb + 16);
    float4 q3 = *(const float4*)(row + kb + 20);
    v16bf a;
    a[0]  = (__bf16)q0.x; a[1]  = (__bf16)q0.y; a[2]  = (__bf16)q0.z; a[3]  = (__bf16)q0.w;
    a[4]  = (__bf16)q1.x; a[5]  = (__bf16)q1.y; a[6]  = (__bf16)q1.z; a[7]  = (__bf16)q1.w;
    a[8]  = (__bf16)q2.x; a[9]  = (__bf16)q2.y; a[10] = (__bf16)q2.z; a[11] = (__bf16)q2.w;
    a[12] = (__bf16)q3.x; a[13] = (__bf16)q3.y; a[14] = (__bf16)q3.z; a[15] = (__bf16)q3.w;
#pragma unroll
    for (int t = 0; t < NTILES; ++t) {
      v16bf b = *(const v16bf*)(Bs + (((kt * NTILES + t) << 9) + (lane << 4)));
      acc[t] = __builtin_amdgcn_wmma_f32_16x16x32_bf16(
          false, a, false, b, (short)0, acc[t], false, false);
    }
  }
  // partial K tail (guarded scalar loads), outside the hot loop
  for (int kt = kfull; kt < KTILES; ++kt) {
    const int kb = (kt << 5) + (kh << 3);
    v16bf a;
#pragma unroll
    for (int j = 0; j < 8; ++j) { int k = kb + j;      a[j]     = (__bf16)(k < K ? row[k] : 0.f); }
#pragma unroll
    for (int j = 0; j < 8; ++j) { int k = kb + 16 + j; a[8 + j] = (__bf16)(k < K ? row[k] : 0.f); }
#pragma unroll
    for (int t = 0; t < NTILES; ++t) {
      v16bf b = *(const v16bf*)(Bs + (((kt * NTILES + t) << 9) + (lane << 4)));
      acc[t] = __builtin_amdgcn_wmma_f32_16x16x32_bf16(
          false, a, false, b, (short)0, acc[t], false, false);
    }
  }
  // C layout: VGPR j -> M = m0 + 8*kh + j, N = t*16 + (lane&15)
  const int ldc = NTILES * 16;
#pragma unroll
  for (int t = 0; t < NTILES; ++t)
#pragma unroll
    for (int j = 0; j < 8; ++j)
      C[(size_t)(m0 + (kh << 3) + j) * ldc + t * 16 + r] = acc[t][j];
}

// ---------------------------------------------------------------------------
// Elementwise / graph kernels
// ---------------------------------------------------------------------------
__global__ void fill_f32(float* p, float v, int n) {
  int i = blockIdx.x * blockDim.x + threadIdx.x;
  if (i < n) p[i] = v;
}
__global__ void fill_u32(unsigned* p, unsigned v, int n) {
  int i = blockIdx.x * blockDim.x + threadIdx.x;
  if (i < n) p[i] = v;
}

// el[n] = h[n,:].al ; er[n] = h[n,:].ar
__global__ void dot_el_er(const float* __restrict__ H, const float* __restrict__ al,
                          const float* __restrict__ ar, float* __restrict__ el,
                          float* __restrict__ er, int N, int F, int ldh) {
  int n = blockIdx.x * blockDim.x + threadIdx.x;
  if (n >= N) return;
  const float* row = H + (size_t)n * ldh;
  float a = 0.f, b = 0.f;
  for (int f = 0; f < F; ++f) {
    float h = row[f];
    a = fmaf(h, al[f], a);
    b = fmaf(h, ar[f], b);
  }
  el[n] = a;
  er[n] = b;
}

// e = leaky_relu(el[src]+er[dst]); segment max via ordered-uint atomicMax
__global__ void edge_score(const int* __restrict__ src, const int* __restrict__ dst,
                           const float* __restrict__ el, const float* __restrict__ er,
                           float* __restrict__ e, unsigned* __restrict__ m, int E) {
  int i = blockIdx.x * blockDim.x + threadIdx.x;
  if (i >= E) return;
  float v = el[src[i]] + er[dst[i]];
  v = v > 0.f ? v : 0.2f * v;          // NEG_SLOPE = 0.2
  e[i] = v;
  atomicMax(m + dst[i], floatOrd(v));
}

// ex = exp(e - m[dst]); segment sum s[dst] += ex  (overwrites e with ex)
__global__ void edge_exp(const int* __restrict__ dst, float* __restrict__ e,
                         const unsigned* __restrict__ m, float* __restrict__ s, int E) {
  int i = blockIdx.x * blockDim.x + threadIdx.x;
  if (i >= E) return;
  int d = dst[i];
  float ex = __expf(e[i] - ordFloat(m[d]));
  e[i] = ex;
  atomicAdd(s + d, ex);
}

// Out[dst,:] += (ex/s[dst]) * H[src,:]  — one wave per edge, lanes = features
__global__ void edge_aggregate(const int* __restrict__ src, const int* __restrict__ dst,
                               const float* __restrict__ ex, const float* __restrict__ s,
                               const float* __restrict__ H, float* __restrict__ Out,
                               int E, int F, int ldh, int ldo) {
  int wave = (int)((blockIdx.x * blockDim.x + threadIdx.x) >> 5);
  int lane = threadIdx.x & 31;
  if (wave >= E) return;
  int sN = src[wave], dN = dst[wave];
  float alpha = ex[wave] / s[dN];
  for (int f = lane; f < F; f += 32)
    atomicAdd(Out + (size_t)dN * ldo + f, alpha * H[(size_t)sN * ldh + f]);
}

// Half-wave-per-edge variant for F <= 16 (layer 2): no idle lanes
__global__ void edge_aggregate_h(const int* __restrict__ src, const int* __restrict__ dst,
                                 const float* __restrict__ ex, const float* __restrict__ s,
                                 const float* __restrict__ H, float* __restrict__ Out,
                                 int E, int F, int ldh, int ldo) {
  int half = (int)((blockIdx.x * blockDim.x + threadIdx.x) >> 4);
  int lane = threadIdx.x & 15;
  if (half >= E) return;
  int sN = src[half], dN = dst[half];
  float alpha = ex[half] / s[dN];
  if (lane < F)
    atomicAdd(Out + (size_t)dN * ldo + lane, alpha * H[(size_t)sN * ldh + lane]);
}

// agg[n,f] = ELU(agg[n,f] + b[f]) for f<F ; zero the pad columns f in [F,ldo)
__global__ void bias_elu_pad(float* __restrict__ agg, const float* __restrict__ b,
                             int N, int F, int ldo) {
  int i = blockIdx.x * blockDim.x + threadIdx.x;
  int n = i / ldo, f = i - n * ldo;
  if (n >= N) return;
  float v = 0.f;
  if (f < F) {
    v = agg[i] + b[f];
    v = v > 0.f ? v : expm1f(v);       // jax.nn.elu, alpha=1
  }
  agg[i] = v;
}

__global__ void bias_add(float* __restrict__ out, const float* __restrict__ b,
                         int N, int F) {
  int i = blockIdx.x * blockDim.x + threadIdx.x;
  if (i >= N * F) return;
  out[i] += b[i % F];
}

// ---------------------------------------------------------------------------
// Launcher
// ---------------------------------------------------------------------------
static inline int cdiv(int a, int b) { return (a + b - 1) / b; }

extern "C" void kernel_launch(void* const* d_in, const int* in_sizes, int n_in,
                              void* d_out, int out_size, void* d_ws, size_t ws_size,
                              hipStream_t stream) {
  (void)in_sizes; (void)n_in; (void)out_size; (void)ws_size;
  const float* features = (const float*)d_in[0];
  const int*   src      = (const int*)d_in[1];
  const int*   dst      = (const int*)d_in[2];
  const float* W1  = (const float*)d_in[3];
  const float* al1 = (const float*)d_in[4];
  const float* ar1 = (const float*)d_in[5];
  const float* b1  = (const float*)d_in[6];
  const float* W2  = (const float*)d_in[7];
  const float* al2 = (const float*)d_in[8];
  const float* ar2 = (const float*)d_in[9];
  const float* b2  = (const float*)d_in[10];
  float* out = (float*)d_out;

  const int N = 100000, E = 3200000;
  const int F_IN = 500, F_HID = 41, N_CLS = 16;
  const int NP1 = 48;    // padded hidden width: 3 WMMA n-tiles, also agg stride
  const int KP1 = 512;   // F_IN padded to k-tile multiple (32) -> 16 k-tiles
  const int KP2 = 64;    // NP1(=48) padded -> 2 k-tiles

  // Workspace carve-up (~52 MB, 256B aligned blocks)
  char* ws = (char*)d_ws;
  size_t off = 0;
  auto alloc = [&](size_t bytes) -> void* {
    off = (off + 255) & ~(size_t)255;
    void* p = ws + off;
    off += bytes;
    return p;
  };
  __bf16*  W1pk = (__bf16*)alloc((size_t)(KP1 / 32) * (NP1 / 16) * 512 * 2);
  __bf16*  W2pk = (__bf16*)alloc((size_t)(KP2 / 32) * 1 * 512 * 2);
  float*   hbuf = (float*)alloc((size_t)N * NP1 * 4);   // h1 proj, reused for h2
  float*   agg  = (float*)alloc((size_t)N * NP1 * 4);   // layer-1 aggregate
  float*   el   = (float*)alloc((size_t)N * 4);
  float*   er   = (float*)alloc((size_t)N * 4);
  unsigned* m   = (unsigned*)alloc((size_t)N * 4);
  float*   s    = (float*)alloc((size_t)N * 4);
  float*   ebuf = (float*)alloc((size_t)E * 4);

  const int TB = 256;

  // ---- weight packing (every call; capture-safe, deterministic) ----
  pack_w_bf16<<<cdiv((KP1 / 32) * (NP1 / 16) * 512, TB), TB, 0, stream>>>(
      W1, F_IN, F_HID, KP1, NP1, W1pk);
  pack_w_bf16<<<cdiv((KP2 / 32) * 1 * 512, TB), TB, 0, stream>>>(
      W2, F_HID, N_CLS, KP2, 16, W2pk);

  // ================= layer 1 =================
  // h1 = features @ W1  (bf16 WMMA, f32 accum, B staged in LDS) -> hbuf[N,48]
  wmma_gemm_bf16<3, 16><<<cdiv(N / 16, 8), 256, 0, stream>>>(
      features, W1pk, hbuf, N, F_IN, F_IN);
  dot_el_er<<<cdiv(N, TB), TB, 0, stream>>>(hbuf, al1, ar1, el, er, N, F_HID, NP1);

  fill_u32<<<cdiv(N, TB), TB, 0, stream>>>(m, ORD_NEG_INF, N);
  fill_f32<<<cdiv(N, TB), TB, 0, stream>>>(s, 0.f, N);
  fill_f32<<<cdiv(N * NP1, TB), TB, 0, stream>>>(agg, 0.f, N * NP1);

  edge_score<<<cdiv(E, TB), TB, 0, stream>>>(src, dst, el, er, ebuf, m, E);
  edge_exp<<<cdiv(E, TB), TB, 0, stream>>>(dst, ebuf, m, s, E);
  edge_aggregate<<<E / (TB / 32), TB, 0, stream>>>(src, dst, ebuf, s, hbuf, agg,
                                                   E, F_HID, NP1, NP1);
  // +b1, ELU, and zero the pad columns so GEMM2 can use K=48 directly
  bias_elu_pad<<<cdiv(N * NP1, TB), TB, 0, stream>>>(agg, b1, N, F_HID, NP1);

  // ================= layer 2 =================
  // h2 = agg @ W2 -> hbuf[N,16]  (pad cols of agg are zero, so K=48 is exact)
  wmma_gemm_bf16<1, 2><<<cdiv(N / 16, 8), 256, 0, stream>>>(
      agg, W2pk, hbuf, N, NP1, NP1);
  dot_el_er<<<cdiv(N, TB), TB, 0, stream>>>(hbuf, al2, ar2, el, er, N, N_CLS, 16);

  fill_u32<<<cdiv(N, TB), TB, 0, stream>>>(m, ORD_NEG_INF, N);
  fill_f32<<<cdiv(N, TB), TB, 0, stream>>>(s, 0.f, N);
  fill_f32<<<cdiv(N * N_CLS, TB), TB, 0, stream>>>(out, 0.f, N * N_CLS);

  edge_score<<<cdiv(E, TB), TB, 0, stream>>>(src, dst, el, er, ebuf, m, E);
  edge_exp<<<cdiv(E, TB), TB, 0, stream>>>(dst, ebuf, m, s, E);
  edge_aggregate_h<<<E / (TB / 16), TB, 0, stream>>>(src, dst, ebuf, s, hbuf, out,
                                                     E, N_CLS, 16, N_CLS);
  bias_add<<<cdiv(N * N_CLS, TB), TB, 0, stream>>>(out, b2, N, N_CLS);
}
